// TriangleAttentionStartingNode_91207925498263
// MI455X (gfx1250) — matrix operand order repaired and checked by reading, not compiled, not run on hardware
//
#include <hip/hip_runtime.h>
#include <hip/hip_bf16.h>
#include <math.h>
#include <stdint.h>

// ---------------------------------------------------------------------------
// Triangle attention (starting node), MI455X / gfx1250, wave32 + WMMA bf16.
// Pipeline (bf16 activations in workspace, f32 accumulation in WMMA):
//   1. ln_kernel   : pair -> pn (bf16)                      [65536 x 128]
//   2. proj5_gemm  : pn @ {Wq,Wk,Wv,Wg} -> bf16 x4, pn @ Wb -> bias (WMMA)
//   3. attn_kernel : per (i,h) attention (TDM load of V tile) + sigmoid gate
//   4. out_gemm    : o_gated @ Wout -> d_out (f32)
// ---------------------------------------------------------------------------

typedef __attribute__((ext_vector_type(16))) __bf16 bf16x16;
typedef __attribute__((ext_vector_type(8)))  float  f32x8;
typedef __attribute__((ext_vector_type(4)))  unsigned int u32x4;
typedef __attribute__((ext_vector_type(8)))  int    i32x8;
typedef __attribute__((ext_vector_type(4)))  int    i32x4;
typedef unsigned short u16;

#if __has_builtin(__builtin_amdgcn_tensor_load_to_lds)
#define HAVE_TDM 1
#else
#define HAVE_TDM 0
#endif

union FragAB {
    uint4   q[2];     // two b128 chunks = 16 bf16 values
    bf16x16 b;
};
static_assert(sizeof(FragAB) == 32, "frag size");

__device__ __forceinline__ u16 f2bfu(float f) {
    unsigned u = __builtin_bit_cast(unsigned, f);
    unsigned r = u + 0x7fffu + ((u >> 16) & 1u);   // round-to-nearest-even
    return (u16)(r >> 16);
}
__device__ __forceinline__ float bf2f(u16 s) {
    unsigned u = ((unsigned)s) << 16;
    return __builtin_bit_cast(float, u);
}

#define WMMA_BF16(A, B, C) \
    __builtin_amdgcn_wmma_f32_16x16x32_bf16(false, (A), false, (B), (short)0, (C), false, false)

// ---------------------------------------------------------------------------
// Kernel 1: LayerNorm over 128 channels. One wave32 per row, 4 ch/lane.
// ---------------------------------------------------------------------------
__global__ __launch_bounds__(256) void ln_kernel(const float* __restrict__ pair,
                                                 const float* __restrict__ gamma,
                                                 const float* __restrict__ beta,
                                                 u16* __restrict__ pn) {
    const int lane = threadIdx.x & 31;
    const int row  = blockIdx.x * 8 + (threadIdx.x >> 5);   // 65536 rows

    float4 x = ((const float4*)pair)[(size_t)row * 32 + lane];
    float s  = x.x + x.y + x.z + x.w;
    float sq = x.x * x.x + x.y * x.y + x.z * x.z + x.w * x.w;
#pragma unroll
    for (int m = 1; m < 32; m <<= 1) {
        s  += __shfl_xor(s,  m, 32);
        sq += __shfl_xor(sq, m, 32);
    }
    const float mean = s * (1.0f / 128.0f);
    const float var  = sq * (1.0f / 128.0f) - mean * mean;
    const float rstd = rsqrtf(var + 1e-5f);

    float4 g4 = ((const float4*)gamma)[lane];
    float4 b4 = ((const float4*)beta)[lane];
    float y0 = (x.x - mean) * rstd * g4.x + b4.x;
    float y1 = (x.y - mean) * rstd * g4.y + b4.y;
    float y2 = (x.z - mean) * rstd * g4.z + b4.z;
    float y3 = (x.w - mean) * rstd * g4.w + b4.w;

    unsigned long long p = (unsigned long long)f2bfu(y0)
                         | ((unsigned long long)f2bfu(y1) << 16)
                         | ((unsigned long long)f2bfu(y2) << 32)
                         | ((unsigned long long)f2bfu(y3) << 48);
    ((unsigned long long*)pn)[(size_t)row * 32 + lane] = p;
}

// ---------------------------------------------------------------------------
// Kernel 2: fused 5-way projection GEMM via WMMA.
//   pn [65536 x 128] bf16, A-fragments loaded once and reused across all
//   weight matrices. Weights staged transposed (bf16 [n][k]) in one 32 KB LDS
//   buffer; Wb padded to a 16-column tile (cols 4..15 zero) so the pairwise
//   bias also runs on WMMA; its f32 result is stored as bias[h][j*256+k].
// ---------------------------------------------------------------------------
__global__ __launch_bounds__(256) void proj5_gemm(
    const u16*   __restrict__ pn,
    const float* __restrict__ Wq, const float* __restrict__ Wk,
    const float* __restrict__ Wv, const float* __restrict__ Wg,
    const float* __restrict__ Wb,
    u16* __restrict__ q, u16* __restrict__ k, u16* __restrict__ v,
    u16* __restrict__ g, float* __restrict__ bias_h) {
    __shared__ u16 Wt[128 * 128];   // 32 KB, [n][k], refilled per weight

    const int lane = threadIdx.x & 31;
    const int wave = threadIdx.x >> 5;
    const int M0   = blockIdx.x * 128 + wave * 16;
    const int hi   = (lane >= 16);
    const int l15  = lane & 15;

    // Preload A fragments for all four 32-wide K chunks (kept in VGPRs).
    FragAB afr[4];
#pragma unroll
    for (int kk = 0; kk < 4; ++kk) {
        const uint4* ap =
            (const uint4*)(pn + (size_t)(M0 + l15) * 128 + kk * 32 + hi * 16);
        afr[kk].q[0] = ap[0];
        afr[kk].q[1] = ap[1];   // +8 bf16
    }

    const float* Ws[4] = {Wq, Wk, Wv, Wg};
    u16*         Ds[4] = {q, k, v, g};

    for (int w = 0; w < 5; ++w) {
        // --- stage weight (transposed, bf16) ---
        if (w < 4) {
            const float* W = Ws[w];
            for (int idx = threadIdx.x; idx < 128 * 128; idx += 256) {
                int kk = idx >> 7, n = idx & 127;
                Wt[n * 128 + kk] = f2bfu(W[idx]);
            }
        } else {
            for (int idx = threadIdx.x; idx < 16 * 128; idx += 256) {
                int n = idx >> 7, kk = idx & 127;
                float val = (n < 4) ? Wb[kk * 4 + n] : 0.0f;
                Wt[n * 128 + kk] = f2bfu(val);
            }
        }
        __syncthreads();

        if (w < 4) {
            f32x8 acc[8];
#pragma unroll
            for (int nt = 0; nt < 8; ++nt)
                acc[nt] = (f32x8){0.f, 0.f, 0.f, 0.f, 0.f, 0.f, 0.f, 0.f};
#pragma unroll
            for (int kk = 0; kk < 4; ++kk) {
#pragma unroll
                for (int nt = 0; nt < 8; ++nt) {
                    FragAB b;
                    const uint4* bp =
                        (const uint4*)(&Wt[(nt * 16 + l15) * 128 + kk * 32 + hi * 16]);
                    b.q[0] = bp[0];
                    b.q[1] = bp[1];
                    acc[nt] = WMMA_BF16(afr[kk].b, b.b, acc[nt]);
                }
            }
            u16* dst = Ds[w];
#pragma unroll
            for (int nt = 0; nt < 8; ++nt) {
                const int col   = nt * 16 + l15;
                const int mbase = M0 + hi * 8;
#pragma unroll
                for (int r = 0; r < 8; ++r)
                    dst[(size_t)(mbase + r) * 128 + col] = f2bfu(acc[nt][r]);
            }
        } else {
            // pairwise bias: single 16-wide n-tile, only cols 0..3 valid
            f32x8 acc = (f32x8){0.f, 0.f, 0.f, 0.f, 0.f, 0.f, 0.f, 0.f};
#pragma unroll
            for (int kk = 0; kk < 4; ++kk) {
                FragAB b;
                const uint4* bp = (const uint4*)(&Wt[l15 * 128 + kk * 32 + hi * 16]);
                b.q[0] = bp[0];
                b.q[1] = bp[1];
                acc = WMMA_BF16(afr[kk].b, b.b, acc);
            }
            const int col   = l15;       // head index
            const int mbase = M0 + hi * 8;
            if (col < 4) {
#pragma unroll
                for (int r = 0; r < 8; ++r)
                    bias_h[(size_t)col * 65536 + (mbase + r)] = acc[r];
            }
        }
        __syncthreads();
    }
}

// ---------------------------------------------------------------------------
// Kernel 3: attention per (i, head).  S = Q K^T (WMMA) ; softmax_k ; O = P V
// (WMMA via LDS round-trip) ; O *= sigmoid(g).  V tile (256x32 bf16, row
// stride 128) is DMA'd to LDS by the Tensor Data Mover when available.
// ---------------------------------------------------------------------------
__global__ __launch_bounds__(256) void attn_kernel(const u16*   __restrict__ q,
                                                   const u16*   __restrict__ k,
                                                   const u16*   __restrict__ v,
                                                   const u16*   __restrict__ g,
                                                   const float* __restrict__ bias_h,
                                                   u16*         __restrict__ o) {
    __shared__ u16 v_s[256 * 32];    // 16 KB: V tile row-major [k][c] (TDM dest)
    __shared__ u16 vT[32 * 256];     // 16 KB: V transposed [c][k]
    __shared__ u16 ps[8 * 16 * 32];  //  8 KB: per-wave P staging (bf16)

    const int i = blockIdx.x >> 2;
    const int h = blockIdx.x & 3;
    const int lane = threadIdx.x & 31;
    const int wave = threadIdx.x >> 5;
    const int hi   = (lane >= 16);
    const int l15  = lane & 15;

#if HAVE_TDM
    // --- TDM: async tensor load of V tile (2D, 2-byte elems) into LDS ---
    if (wave == 0) {
        uint64_t ga = (uint64_t)(uintptr_t)(v + (size_t)(i * 256) * 128 + h * 32);
        unsigned  lds_off = (unsigned)(uintptr_t)(&v_s[0]);
        // D# group0: count=1 | lds_addr | global_addr[56:0] | type=2
        u32x4 g0 = {1u, lds_off, (unsigned)ga,
                    ((unsigned)(ga >> 32) & 0x01FFFFFFu) | 0x80000000u};
        // D# group1: data_size=2B; tensor_dim0=32, dim1=256; tile 32x256;
        //            tensor_dim0_stride=128 elements.
        i32x8 g1 = {0x00010000, (int)(32u << 16), (int)(256u << 16),
                    (int)(32u << 16), 256, 128, 0, 0};
        i32x4 gz = {0, 0, 0, 0};
#if __clang_major__ >= 23
        i32x8 gz8 = {0, 0, 0, 0, 0, 0, 0, 0};
        __builtin_amdgcn_tensor_load_to_lds(g0, g1, gz, gz, gz8, 0);
#else
        __builtin_amdgcn_tensor_load_to_lds(g0, g1, gz, gz, 0);
#endif
#if __has_builtin(__builtin_amdgcn_s_wait_tensorcnt)
        __builtin_amdgcn_s_wait_tensorcnt(0);
#else
        asm volatile("s_wait_tensorcnt 0x0" ::: "memory");
#endif
    }
    __syncthreads();
    // Transpose from LDS v_s into vT: thread = k-row.
    {
        const int krow = threadIdx.x;
        const uint4* vp = (const uint4*)(v_s + krow * 32);
#pragma unroll
        for (int cc = 0; cc < 4; ++cc) {
            uint4 d = vp[cc];
            const u16* e = (const u16*)&d;
#pragma unroll
            for (int x = 0; x < 8; ++x) vT[(cc * 8 + x) * 256 + krow] = e[x];
        }
    }
#else
    // Fallback: stage V transposed straight from global.
    {
        const int krow = threadIdx.x;
        const uint4* vp = (const uint4*)(v + (size_t)(i * 256 + krow) * 128 + h * 32);
#pragma unroll
        for (int cc = 0; cc < 4; ++cc) {
            uint4 d = vp[cc];
            const u16* e = (const u16*)&d;
#pragma unroll
            for (int x = 0; x < 8; ++x) vT[(cc * 8 + x) * 256 + krow] = e[x];
        }
    }
#endif
    __syncthreads();

    const float scale = 0.17677669529663687f;   // 1/sqrt(32)
    u16* pw = ps + wave * 16 * 32;

    for (int rep = 0; rep < 2; ++rep) {
        const int J0 = (wave + rep * 8) * 16;

        // Q fragment (A-matrix, M=j-tile, K=c=32), direct from global.
        FragAB qa;
        {
            const uint4* qp =
                (const uint4*)(q + (size_t)(i * 256 + J0 + l15) * 128 + h * 32 + hi * 16);
            qa.q[0] = qp[0];
            qa.q[1] = qp[1];
        }

        // S = Q K^T over all 16 k-tiles.
        f32x8 s_acc[16];
#pragma unroll
        for (int kt = 0; kt < 16; ++kt)
            s_acc[kt] = (f32x8){0.f, 0.f, 0.f, 0.f, 0.f, 0.f, 0.f, 0.f};
#pragma unroll
        for (int kt = 0; kt < 16; ++kt) {
            FragAB kb;   // B-matrix: N=k column, K=c
            const uint4* kp = (const uint4*)(k + (size_t)(i * 256 + kt * 16 + l15) * 128 +
                                             h * 32 + hi * 16);
            kb.q[0] = kp[0];
            kb.q[1] = kp[1];
            s_acc[kt] = WMMA_BF16(qa.b, kb.b, s_acc[kt]);
        }

        // scale + pairwise bias b[j,k,h]
        const int jbase = J0 + hi * 8;
#pragma unroll
        for (int kt = 0; kt < 16; ++kt) {
            const int kcol = kt * 16 + l15;
#pragma unroll
            for (int r = 0; r < 8; ++r) {
                float bv = bias_h[(size_t)h * 65536 + (size_t)(jbase + r) * 256 + kcol];
                s_acc[kt][r] = s_acc[kt][r] * scale + bv;
            }
        }

        // softmax over k (16 tiles in-register + 16-lane shuffle reduce)
        float rmax[8], rsum[8];
#pragma unroll
        for (int r = 0; r < 8; ++r) {
            float m = -3.0e38f;
#pragma unroll
            for (int kt = 0; kt < 16; ++kt) m = fmaxf(m, s_acc[kt][r]);
#pragma unroll
            for (int d = 1; d < 16; d <<= 1) m = fmaxf(m, __shfl_xor(m, d, 32));
            rmax[r] = m;
        }
#pragma unroll
        for (int r = 0; r < 8; ++r) {
            float sum = 0.f;
#pragma unroll
            for (int kt = 0; kt < 16; ++kt) {
                float e = __expf(s_acc[kt][r] - rmax[r]);
                s_acc[kt][r] = e;
                sum += e;
            }
#pragma unroll
            for (int d = 1; d < 16; d <<= 1) sum += __shfl_xor(sum, d, 32);
            rsum[r] = 1.0f / sum;
        }
#pragma unroll
        for (int kt = 0; kt < 16; ++kt)
#pragma unroll
            for (int r = 0; r < 8; ++r) s_acc[kt][r] *= rsum[r];

        // O = P V : stage P chunks (16x32 bf16) in per-wave LDS, WMMA with vT.
        f32x8 oacc[2];
        oacc[0] = (f32x8){0.f, 0.f, 0.f, 0.f, 0.f, 0.f, 0.f, 0.f};
        oacc[1] = (f32x8){0.f, 0.f, 0.f, 0.f, 0.f, 0.f, 0.f, 0.f};
#pragma unroll
        for (int kb2 = 0; kb2 < 8; ++kb2) {
#pragma unroll
            for (int tt = 0; tt < 2; ++tt) {
                const int t = kb2 * 2 + tt;
#pragma unroll
                for (int r = 0; r < 8; ++r) {
                    const int m = r + hi * 8;
                    pw[m * 32 + tt * 16 + l15] = f2bfu(s_acc[t][r]);
                }
            }
            asm volatile("s_wait_dscnt 0" ::: "memory");   // same-wave DS RAW
            FragAB pa;   // A-matrix: M=j, K = local k chunk of 32
            {
                const uint4* pp = (const uint4*)(pw + l15 * 32 + hi * 16);
                pa.q[0] = pp[0];
                pa.q[1] = pp[1];
            }
#pragma unroll
            for (int ct = 0; ct < 2; ++ct) {
                FragAB vb;   // B-matrix: N=c column, K=k chunk
                const uint4* vp2 =
                    (const uint4*)(vT + (ct * 16 + l15) * 256 + kb2 * 32 + hi * 16);
                vb.q[0] = vp2[0];
                vb.q[1] = vp2[1];
                oacc[ct] = WMMA_BF16(pa.b, vb.b, oacc[ct]);
            }
        }

        // sigmoid gate + store bf16
#pragma unroll
        for (int ct = 0; ct < 2; ++ct) {
            const int c = ct * 16 + l15;
#pragma unroll
            for (int r = 0; r < 8; ++r) {
                const int j = jbase + r;
                size_t idx = (size_t)(i * 256 + j) * 128 + h * 32 + c;
                float gv  = bf2f(g[idx]);
                float sig = 1.0f / (1.0f + __expf(-gv));
                o[idx] = f2bfu(oacc[ct][r] * sig);
            }
        }
    }
}

// ---------------------------------------------------------------------------
// Kernel 4: [65536 x 128] bf16 @ Wout [128 x 128] -> f32 d_out via WMMA.
// ---------------------------------------------------------------------------
__global__ __launch_bounds__(256) void out_gemm(const u16*   __restrict__ A,
                                                const float* __restrict__ W,
                                                float*       __restrict__ dst) {
    __shared__ u16 Wt[128 * 128];   // 32 KB, [n][k]

    for (int idx = threadIdx.x; idx < 128 * 128; idx += 256) {
        int kk = idx >> 7, n = idx & 127;
        Wt[n * 128 + kk] = f2bfu(W[idx]);
    }
    __syncthreads();

    const int lane = threadIdx.x & 31;
    const int wave = threadIdx.x >> 5;
    const int M0   = blockIdx.x * 128 + wave * 16;
    const int hi   = (lane >= 16);
    const int l15  = lane & 15;

    f32x8 acc[8];
#pragma unroll
    for (int nt = 0; nt < 8; ++nt)
        acc[nt] = (f32x8){0.f, 0.f, 0.f, 0.f, 0.f, 0.f, 0.f, 0.f};

#pragma unroll
    for (int kk = 0; kk < 4; ++kk) {
        FragAB a;
        const uint4* ap = (const uint4*)(A + (size_t)(M0 + l15) * 128 + kk * 32 + hi * 16);
        a.q[0] = ap[0];
        a.q[1] = ap[1];
#pragma unroll
        for (int nt = 0; nt < 8; ++nt) {
            FragAB b;
            const uint4* bp =
                (const uint4*)(&Wt[(nt * 16 + l15) * 128 + kk * 32 + hi * 16]);
            b.q[0] = bp[0];
            b.q[1] = bp[1];
            acc[nt] = WMMA_BF16(a.b, b.b, acc[nt]);
        }
    }

#pragma unroll
    for (int nt = 0; nt < 8; ++nt) {
        const int col   = nt * 16 + l15;
        const int mbase = M0 + hi * 8;
#pragma unroll
        for (int r = 0; r < 8; ++r)
            dst[(size_t)(mbase + r) * 128 + col] = acc[nt][r];
    }
}

// ---------------------------------------------------------------------------
// Host-side launch.
// ---------------------------------------------------------------------------
extern "C" void kernel_launch(void* const* d_in, const int* in_sizes, int n_in,
                              void* d_out, int out_size, void* d_ws, size_t ws_size,
                              hipStream_t stream) {
    const float* pair  = (const float*)d_in[0];
    const float* gamma = (const float*)d_in[1];
    const float* beta  = (const float*)d_in[2];
    const float* Wq    = (const float*)d_in[3];
    const float* Wk    = (const float*)d_in[4];
    const float* Wv    = (const float*)d_in[5];
    const float* Wb    = (const float*)d_in[6];
    const float* Wg    = (const float*)d_in[7];
    const float* Wout  = (const float*)d_in[8];
    float* out = (float*)d_out;

    constexpr size_t SZ = (size_t)65536 * 128 * 2;   // one bf16 activation buffer
    char* ws = (char*)d_ws;
    u16*   pn     = (u16*)(ws);
    u16*   qb     = (u16*)(ws + 1 * SZ);
    u16*   kb     = (u16*)(ws + 2 * SZ);
    u16*   vb     = (u16*)(ws + 3 * SZ);
    u16*   gb     = (u16*)(ws + 4 * SZ);
    float* bias_h = (float*)(ws + 5 * SZ);                       // 1 MB
    u16*   ob     = (u16*)(ws + 5 * SZ + (size_t)4 * 65536 * 4); // bf16

    ln_kernel<<<8192, 256, 0, stream>>>(pair, gamma, beta, pn);

    proj5_gemm<<<512, 256, 0, stream>>>(pn, Wq, Wk, Wv, Wg, Wb,
                                        qb, kb, vb, gb, bias_h);

    attn_kernel<<<1024, 256, 0, stream>>>(qb, kb, vb, gb, bias_h, ob);

    out_gemm<<<512, 256, 0, stream>>>(ob, Wout, out);
}
